// CapsNet_30769145708960
// MI455X (gfx1250) — matrix operand order repaired and checked by reading, not compiled, not run on hardware
//
#include <hip/hip_runtime.h>

// ---------------------------------------------------------------------------
// CapsNet forward for MI455X (gfx1250, wave32, WMMA).
//   conv1 / conv2 : implicit-GEMM, v_wmma_f32_16x16x32_bf16 (f32 accum),
//                   weights staged global->LDS via async-to-LDS copies
//   routing       : per-sample workgroup, u_hat cached in LDS (bf16), f32 math
// ---------------------------------------------------------------------------

typedef __attribute__((ext_vector_type(16))) __bf16 v16bf;
typedef __attribute__((ext_vector_type(8)))  float  v8f;
typedef __attribute__((__vector_size__(16))) int    v4i;

#define AS1 __attribute__((address_space(1)))
#define AS3 __attribute__((address_space(3)))

__device__ __forceinline__ float bf2f(unsigned short h) {
    return __uint_as_float(((unsigned)h) << 16);
}
__device__ __forceinline__ unsigned short f2bf(float f) {
    unsigned u = __float_as_uint(f);
    unsigned r = u + 0x7FFFu + ((u >> 16) & 1u);   // round-to-nearest-even
    return (unsigned short)(r >> 16);
}

// 16-byte global -> LDS copy. Uses the gfx1250 async-to-LDS path when the
// builtin is available (tracked with ASYNCcnt), else a plain vector copy.
__device__ __forceinline__ void copy16_g2l(const void* g, void* l) {
#if __has_builtin(__builtin_amdgcn_global_load_async_to_lds_b128)
    __builtin_amdgcn_global_load_async_to_lds_b128((AS1 v4i*)g, (AS3 v4i*)l, 0, 0);
#else
    ((uint4*)l)[0] = ((const uint4*)g)[0];
#endif
}
__device__ __forceinline__ void wait_async0() {
#if __has_builtin(__builtin_amdgcn_s_wait_asynccnt)
    __builtin_amdgcn_s_wait_asynccnt(0);
#elif defined(__gfx1250__)
    asm volatile("s_wait_asynccnt 0x0" ::: "memory");
#endif
}

// --------------------------- f32 -> bf16 convert ---------------------------
__global__ void f32_to_bf16_kernel(const float* __restrict__ src,
                                   unsigned short* __restrict__ dst, int n) {
    int i = blockIdx.x * blockDim.x + threadIdx.x;
    if (i < n) dst[i] = f2bf(src[i]);
}

// ------------------------- implicit-GEMM conv (WMMA) -----------------------
// GEMM view: M = B*OH*OW (rows = (b,oy,ox)), N = COUT = NT*16, K = CIN*9*9.
// K index kk decomposes as c = kk/81, r = kk%81, ky = r/9, kx = r%9, matching
// the flat weight layout w[o][c][ky][kx] -> wgt[n*KTOT + kk].
// Weights are staged once per block into LDS (async copy), B fragments are
// then built from LDS. MODE 0: bf16 NCHW store (conv1). MODE 1: f32 capsule
// layout u[b][i][c] with i = (ch&3)*36 + sp, c = ch>>2 (conv2).
template<int CIN, int H, int W, int OH, int OW, int STRIDE, int KTOT, int NT, int MODE>
__global__ __launch_bounds__(128)
void conv_wmma_kernel(const unsigned short* __restrict__ act,
                      const unsigned short* __restrict__ wgt,
                      const float* __restrict__ bias,
                      unsigned short* __restrict__ out_bf,
                      float* __restrict__ out_f,
                      int mtiles) {
    __shared__ __align__(16) unsigned short wlds[NT * 16 * KTOT];

    const int lane  = threadIdx.x & 31;          // wave32
    const int wave  = threadIdx.x >> 5;
    const int mtile = blockIdx.x * 4 + wave;     // one 16-row M tile per wave

    // ---- stage full weight matrix into LDS (16B chunks, async) ----
    constexpr int WCHUNKS = (NT * 16 * KTOT * 2) / 16;   // bytes / 16
    for (int c = threadIdx.x; c < WCHUNKS; c += 128)
        copy16_g2l(wgt + c * 8, wlds + c * 8);
    wait_async0();
    __syncthreads();

    if (mtile >= mtiles) return;                 // wave-uniform: EXEC stays full

    const int  mrow  = lane & 15;                // A-matrix M row for this lane
    const int  khalf = lane >> 4;                // K sub-range selector
    const int  gm    = mtile * 16 + mrow;
    const int  bimg  = gm / (OH * OW);
    const int  sp    = gm % (OH * OW);
    const int  oy    = sp / OW;
    const int  ox    = sp % OW;
    const long abase = (long)bimg * CIN * H * W;

    __builtin_prefetch(act + abase, 0, 3);       // pull this image's plane

    v8f acc[NT];
    #pragma unroll
    for (int t = 0; t < NT; ++t)
        #pragma unroll
        for (int j = 0; j < 8; ++j) acc[t][j] = 0.0f;

    const int KT = (KTOT + 31) / 32;
    for (int kt = 0; kt < KT; ++kt) {
        // ---- A fragment: 16x32 bf16, ISA layout ----
        // VGPR v, pair p: K = (v>=4?16:0) + khalf*8 + (v&3)*2 + p
        union { v16bf v; unsigned short us[16]; } A;
        #pragma unroll
        for (int e = 0; e < 16; ++e) {
            const int vreg = e >> 1, p = e & 1;
            const int kl = ((vreg >= 4) ? 16 : 0) + khalf * 8 + (vreg & 3) * 2 + p;
            const int kk = kt * 32 + kl;
            unsigned short val = 0;
            if (kk < KTOT) {
                const int c  = kk / 81;
                const int r  = kk % 81;
                const int ky = r / 9, kx = r % 9;
                const int iy = oy * STRIDE + ky;
                const int ix = ox * STRIDE + kx;
                val = act[abase + ((long)c * H + iy) * W + ix];
            }
            A.us[e] = val;
        }

        // ---- B fragments from LDS: 32x16 bf16 per N tile ----
        // VGPR v, pair p: K = (v>=4?16:0) + khalf*8 + (v&3)*2 + p ; N = lane&15
        #pragma unroll
        for (int t = 0; t < NT; ++t) {
            union { v16bf v; unsigned short us[16]; } Bf;
            const int n = t * 16 + (lane & 15);
            #pragma unroll
            for (int e = 0; e < 16; ++e) {
                const int vreg = e >> 1, p = e & 1;
                const int kl = ((vreg >= 4) ? 16 : 0) + khalf * 8 + (vreg & 3) * 2 + p;
                const int kk = kt * 32 + kl;
                Bf.us[e] = (kk < KTOT) ? wlds[n * KTOT + kk] : (unsigned short)0;
            }
            acc[t] = __builtin_amdgcn_wmma_f32_16x16x32_bf16(
                         false, A.v, false, Bf.v, (short)0, acc[t], false, false);
        }
    }

    // ---- epilogue: C/D layout VGPR r -> M = r + 8*khalf, N = lane&15 ----
    #pragma unroll
    for (int t = 0; t < NT; ++t) {
        const int n = t * 16 + (lane & 15);
        const float bv = bias[n];
        #pragma unroll
        for (int r = 0; r < 8; ++r) {
            const int m   = r + khalf * 8;
            const int row = mtile * 16 + m;
            const int bb  = row / (OH * OW);
            const int spp = row % (OH * OW);
            const float val = acc[t][r] + bv;
            if (MODE == 0) {
                const int oyy = spp / OW, oxx = spp % OW;
                out_bf[(((long)bb * (NT * 16) + n) * OH + oyy) * OW + oxx] = f2bf(val);
            } else {
                const int cc = n >> 2;            // capsule dim
                const int g  = n & 3;
                const int i  = g * 36 + spp;      // primary-capsule index
                out_f[((long)bb * 144 + i) * 8 + cc] = val;
            }
        }
    }
}

// ----------------------------- routing kernel ------------------------------
#define NIN  144
#define NOUT 10
#define DOUT 16

__global__ __launch_bounds__(256)
void routing_kernel(const float* __restrict__ u,     // [B,144,8]
                    const float* __restrict__ Wr,    // [144,10,16,8]
                    float* __restrict__ pred,        // [B,10]
                    float* __restrict__ vout) {      // [B,10,16]
    __shared__ __align__(16) unsigned short uh[NIN * NOUT * DOUT]; // bf16, 46080 B
    __shared__ __align__(16) float u_s[NIN * 8];
    __shared__ float blog[NIN * NOUT];
    __shared__ float c_s[NIN * NOUT];
    __shared__ float v_s[NOUT * DOUT];
    __shared__ float f_s[NOUT];

    const int b   = blockIdx.x;
    const int tid = threadIdx.x;

    // stage u[b] (4608 B = 288 x 16B) via async-to-LDS
    {
        const unsigned short* gsrc = (const unsigned short*)(u + (long)b * NIN * 8);
        for (int c = tid; c < (NIN * 8 * 4) / 16; c += 256)
            copy16_g2l(gsrc + c * 8, ((unsigned short*)u_s) + c * 8);
    }
    for (int idx = tid; idx < NIN * NOUT; idx += 256) blog[idx] = 0.0f;
    wait_async0();
    __syncthreads();

    // u_hat[i, o*16+d] = sum_c W[i,o,d,c] * u[b,i,c]
    for (int idx = tid; idx < NIN * NOUT * DOUT; idx += 256) {
        const int i  = idx / (NOUT * DOUT);
        const int od = idx % (NOUT * DOUT);
        const float* w  = Wr + ((long)i * NOUT * DOUT + od) * 8;
        const float* uu = u_s + i * 8;
        float a = 0.0f;
        #pragma unroll
        for (int c = 0; c < 8; ++c) a += w[c] * uu[c];
        uh[idx] = f2bf(a);
    }
    __syncthreads();

    for (int iter = 0; iter < 3; ++iter) {
        // softmax over out-capsules per input capsule
        for (int i = tid; i < NIN; i += 256) {
            float mx = -1e30f;
            for (int o = 0; o < NOUT; ++o) mx = fmaxf(mx, blog[i * NOUT + o]);
            float e[NOUT], s = 0.0f;
            #pragma unroll
            for (int o = 0; o < NOUT; ++o) { e[o] = expf(blog[i * NOUT + o] - mx); s += e[o]; }
            const float inv = 1.0f / s;
            #pragma unroll
            for (int o = 0; o < NOUT; ++o) c_s[i * NOUT + o] = e[o] * inv;
        }
        __syncthreads();
        // s[o,d] = sum_i c[i,o] * u_hat[i,o,d]
        for (int od = tid; od < NOUT * DOUT; od += 256) {
            const int o = od / DOUT;
            float acc = 0.0f;
            for (int i = 0; i < NIN; ++i)
                acc += c_s[i * NOUT + o] * bf2f(uh[i * NOUT * DOUT + od]);
            v_s[od] = acc;
        }
        __syncthreads();
        // squash factor per out capsule
        for (int o = tid; o < NOUT; o += 256) {
            float sq = 0.0f;
            #pragma unroll
            for (int d = 0; d < DOUT; ++d) { float x = v_s[o * DOUT + d]; sq += x * x; }
            f_s[o] = (sq / (1.0f + sq)) * rsqrtf(sq + 1e-8f);
        }
        __syncthreads();
        for (int od = tid; od < NOUT * DOUT; od += 256) v_s[od] *= f_s[od / DOUT];
        __syncthreads();
        // b_log += <u_hat[i,o,:], v[o,:]>
        for (int idx = tid; idx < NIN * NOUT; idx += 256) {
            const int i = idx / NOUT, o = idx % NOUT;
            float a = 0.0f;
            #pragma unroll
            for (int d = 0; d < DOUT; ++d)
                a += bf2f(uh[(i * NOUT + o) * DOUT + d]) * v_s[o * DOUT + d];
            blog[idx] += a;
        }
        __syncthreads();
    }

    for (int o = tid; o < NOUT; o += 256) {
        float sq = 0.0f;
        #pragma unroll
        for (int d = 0; d < DOUT; ++d) { float x = v_s[o * DOUT + d]; sq += x * x; }
        pred[(long)b * NOUT + o] = sqrtf(sq);
    }
    for (int od = tid; od < NOUT * DOUT; od += 256)
        vout[(long)b * NOUT * DOUT + od] = v_s[od];
}

// ------------------------------- launcher ----------------------------------
extern "C" void kernel_launch(void* const* d_in, const int* in_sizes, int n_in,
                              void* d_out, int out_size, void* d_ws, size_t ws_size,
                              hipStream_t stream) {
    (void)in_sizes; (void)n_in; (void)out_size; (void)ws_size;
    const int B = 2048;

    const float* inp = (const float*)d_in[0];   // [B,1,28,28]
    const float* w1  = (const float*)d_in[1];   // [16,1,9,9]
    const float* b1  = (const float*)d_in[2];   // [16]
    const float* w2  = (const float*)d_in[3];   // [32,16,9,9]
    const float* b2  = (const float*)d_in[4];   // [32]
    const float* Wr  = (const float*)d_in[5];   // [144,10,16,8]

    // workspace carve-out (~39 MB total), 256 B aligned
    char* ws = (char*)d_ws;
    size_t off = 0;
    auto carve = [&](size_t bytes) -> char* {
        char* p = ws + off;
        off = (off + bytes + 255) & ~(size_t)255;
        return p;
    };
    unsigned short* in_bf = (unsigned short*)carve((size_t)B * 784 * 2);
    unsigned short* w1_bf = (unsigned short*)carve((size_t)16 * 81 * 2);
    unsigned short* w2_bf = (unsigned short*)carve((size_t)32 * 16 * 81 * 2);
    unsigned short* act1  = (unsigned short*)carve((size_t)B * 16 * 400 * 2);
    float*          u     = (float*)carve((size_t)B * 144 * 8 * 4);

    const int n0 = B * 784;
    const int n1 = 16 * 81;
    const int n2 = 32 * 16 * 81;
    f32_to_bf16_kernel<<<(n0 + 255) / 256, 256, 0, stream>>>(inp, in_bf, n0);
    f32_to_bf16_kernel<<<(n1 + 255) / 256, 256, 0, stream>>>(w1, w1_bf, n1);
    f32_to_bf16_kernel<<<(n2 + 255) / 256, 256, 0, stream>>>(w2, w2_bf, n2);

    // conv1: [B,1,28,28] -> bf16 [B,16,20,20]   (M=819200, N=16, K=81)
    const int mt1 = (B * 400) / 16;   // 51200 tiles, 4 waves/block
    conv_wmma_kernel<1, 28, 28, 20, 20, 1, 81, 1, 0>
        <<<mt1 / 4, 128, 0, stream>>>(in_bf, w1_bf, b1, act1, nullptr, mt1);

    // conv2: [B,16,20,20] -> f32 u [B,144,8]    (M=73728, N=32, K=1296)
    const int mt2 = (B * 36) / 16;    // 4608 tiles
    conv_wmma_kernel<16, 20, 20, 6, 6, 2, 1296, 2, 1>
        <<<mt2 / 4, 128, 0, stream>>>(act1, w2_bf, b2, nullptr, u, mt2);

    // dynamic routing, one sample per workgroup
    float* pred = (float*)d_out;            // [B,10]
    float* vout = pred + (size_t)B * 10;    // [B,10,16]
    routing_kernel<<<B, 256, 0, stream>>>(u, Wr, pred, vout);
}
